// SDPAttention2D_2259152798578
// MI455X (gfx1250) — matrix operand order repaired and checked by reading, not compiled, not run on hardware
//
#include <hip/hip_runtime.h>

typedef __bf16 bf16;
typedef __attribute__((ext_vector_type(16))) __bf16 v16bf;
typedef __attribute__((ext_vector_type(8)))  __bf16 v8bf;
typedef __attribute__((ext_vector_type(8)))  float   v8f;
typedef __attribute__((ext_vector_type(4)))  unsigned int v4u;
typedef __attribute__((ext_vector_type(8)))  int v8i;
typedef __attribute__((ext_vector_type(4)))  int v4i;

#define NB 2
#define LSEQ 2304
#define CDIM 512
#define NHEAD 8
#define NKV 4
#define HD 64
#define MROWS (NB*LSEQ)   // 4608

static __device__ __forceinline__ v8f wmma_bf16(v16bf a, v16bf b, v8f c) {
  // D = A(16x32 bf16) * B(32x16 bf16) + C(16x16 f32)
  return __builtin_amdgcn_wmma_f32_16x16x32_bf16(false, a, false, b, (short)0, c, false, false);
}

static __device__ __forceinline__ v16bf cat8(v8bf lo, v8bf hi) {
  return __builtin_shufflevector(lo, hi, 0,1,2,3,4,5,6,7,8,9,10,11,12,13,14,15);
}

static __device__ __forceinline__ v8f vzero8() {
  v8f z;
#pragma unroll
  for (int i = 0; i < 8; ++i) z[i] = 0.0f;
  return z;
}

// ---- Tensor Data Mover: 2D bf16 tile (tile_d0 contiguous elems x tile_d1 rows,
// row stride stride0 elems) global -> LDS (packed contiguous). D# per ISA 08 §8.
static __device__ __forceinline__ void tdm_load_tile_bf16(
    const bf16* gsrc, bf16* ldst,
    unsigned tile_d0, unsigned tile_d1,
    unsigned tensor_d0, unsigned tensor_d1,
    unsigned long long stride0_elems) {
  unsigned long long ga = (unsigned long long)(uintptr_t)gsrc;   // byte address
  unsigned lds = (unsigned)(uintptr_t)ldst;                      // low 32 bits = LDS byte addr
  v4u g0;
  g0[0] = 1u;                                                    // count=1, user D#
  g0[1] = lds;                                                   // lds_addr
  g0[2] = (unsigned)(ga & 0xffffffffull);                        // global_addr[31:0]
  g0[3] = (unsigned)((ga >> 32) & 0x1ffffffull) | (2u << 30);    // global_addr[56:32] | type=2
  v8i g1;
  g1[0] = (int)(1u << 16);                                       // wg_mask=0, data_size=1 (2B)
  g1[1] = (int)((tensor_d0 & 0xffffu) << 16);                    // tensor_dim0[15:0]
  g1[2] = (int)((tensor_d0 >> 16) | ((tensor_d1 & 0xffffu) << 16));
  g1[3] = (int)((tensor_d1 >> 16) | ((tile_d0 & 0xffffu) << 16));
  g1[4] = (int)(tile_d1 & 0xffffu);                              // tile_dim1, tile_dim2=0
  g1[5] = (int)(stride0_elems & 0xffffffffull);                  // tensor_dim0_stride[31:0]
  g1[6] = (int)((stride0_elems >> 32) & 0xffffull);              // stride[47:32], dim1_stride=0
  g1[7] = 0;
  v4i gz4 = {0, 0, 0, 0};                                        // 2D: groups 2/3 unused
  v8i gz8 = {0, 0, 0, 0, 0, 0, 0, 0};
  __builtin_amdgcn_tensor_load_to_lds(g0, g1, gz4, gz4, gz8, 0);
}

// ---------------- elementwise casts / weight transpose ----------------

__global__ void k_cast_bf16(const float* __restrict__ s, bf16* __restrict__ d, int n) {
  int i = blockIdx.x * blockDim.x + threadIdx.x;
  if (i < n) d[i] = (bf16)s[i];
}

// src is K x N row-major (f32); dst is N x K row-major (bf16)
__global__ void k_transpose_bf16(const float* __restrict__ s, bf16* __restrict__ d, int K, int N) {
  int i = blockIdx.x * blockDim.x + threadIdx.x;
  if (i < K * N) {
    int k = i / N, n = i % N;
    d[n * K + k] = (bf16)s[i];
  }
}

// ---------------- generic bf16 WMMA GEMM: C(MxN,f32) = A(MxK) * BT(NxK)^T ----------------
// grid = (N/64, M/64), block = 128 (4 waves); wave w: rows m0+w*16, 64 cols.

__global__ __launch_bounds__(128) void k_gemm_wmma(
    const bf16* __restrict__ A, const bf16* __restrict__ BT, float* __restrict__ C,
    int lda, int ldb, int ldc, int K) {
  int lane = threadIdx.x & 31;
  int w    = threadIdx.x >> 5;
  int m0   = blockIdx.y * 64 + w * 16;
  int n0   = blockIdx.x * 64;
  int half = lane >> 4;
  int l16  = lane & 15;
  int koff = half * 8;

  v8f acc0 = vzero8(), acc1 = vzero8(), acc2 = vzero8(), acc3 = vzero8();
  const bf16* arow = A + (m0 + l16) * lda;

  for (int k0 = 0; k0 < K; k0 += 32) {
    v16bf a = *(const v16bf*)(arow + k0 + half * 16);
    v16bf b0 = *(const v16bf*)(BT + (n0 +  0 + l16) * ldb + k0 + half * 16);
    v16bf b1 = *(const v16bf*)(BT + (n0 + 16 + l16) * ldb + k0 + half * 16);
    v16bf b2 = *(const v16bf*)(BT + (n0 + 32 + l16) * ldb + k0 + half * 16);
    v16bf b3 = *(const v16bf*)(BT + (n0 + 48 + l16) * ldb + k0 + half * 16);
    acc0 = wmma_bf16(a, b0, acc0);
    acc1 = wmma_bf16(a, b1, acc1);
    acc2 = wmma_bf16(a, b2, acc2);
    acc3 = wmma_bf16(a, b3, acc3);
  }

#pragma unroll
  for (int r = 0; r < 8; ++r) {
    int grow = m0 + r + half * 8;
    float* cr = C + grow * ldc + n0 + l16;
    cr[0]  = acc0[r];
    cr[16] = acc1[r];
    cr[32] = acc2[r];
    cr[48] = acc3[r];
  }
}

// ---------------- RMSNorm + RoPE + layout pack ----------------
// one wave per (b, l, slot); slots: 0-7 q heads, 8-11 k heads, 12-15 v heads.
// qh: [B,nH,L,hd] bf16 ; kh: [B,nKV,L,hd] bf16 ; vt: [B,nKV,hd,L] bf16 (transposed)

__global__ __launch_bounds__(256) void k_norm_rope_pack(
    const float* __restrict__ qkv, const float* __restrict__ qw, const float* __restrict__ kw,
    bf16* __restrict__ qh, bf16* __restrict__ kh, bf16* __restrict__ vt) {
  int wid  = blockIdx.x * 8 + (threadIdx.x >> 5);
  int lane = threadIdx.x & 31;
  int slot = wid & 15;
  int t    = wid >> 4;
  int l    = t % LSEQ;
  int b    = t / LSEQ;
  const float* row = qkv + (b * LSEQ + l) * 1024;

  if (slot < 12) {
    int h = (slot < 8) ? slot : (slot - 8);
    const float* src   = row + ((slot < 8) ? slot * 64 : 512 + (slot - 8) * 64);
    const float* wnorm = (slot < 8) ? qw : kw;
    float v0 = src[lane], v1 = src[lane + 32];
    float ss = v0 * v0 + v1 * v1;
#pragma unroll
    for (int off = 16; off >= 1; off >>= 1) ss += __shfl_xor(ss, off, 32);
    float rms = rsqrtf(ss * (1.0f / 64.0f) + 1e-6f);
    float n0 = v0 * rms * wnorm[lane];
    float n1 = v1 * rms * wnorm[lane + 32];
    // RoPE: half=32; lane i holds d=i and d=i+32
    float invf = __expf(-logf(10000.0f) * ((float)lane / 32.0f));
    float fr = (float)l * invf;
    float cs = cosf(fr), sn = sinf(fr);
    float o0 = n0 * cs - n1 * sn;
    float o1 = n0 * sn + n1 * cs;
    bf16* dst = (slot < 8)
      ? qh + ((b * NHEAD + h) * LSEQ + l) * 64
      : kh + ((b * NKV   + h) * LSEQ + l) * 64;
    dst[lane]      = (bf16)o0;
    dst[lane + 32] = (bf16)o1;
  } else {
    int h = slot - 12;
    const float* src = row + 768 + h * 64;
    bf16* dst = vt + (b * NKV + h) * 64 * LSEQ;
    dst[lane * LSEQ + l]        = (bf16)src[lane];
    dst[(lane + 32) * LSEQ + l] = (bf16)src[lane + 32];
  }
}

// ---------------- flash attention (online softmax, WMMA bf16, TDM-staged K/V) ----------------
// grid = (L/64, nH, B), block = 128; wave = 16 q rows; key tiles of 32.
// K tile (32 keys x 64 d) and V tile (64 d x 32 keys) are shared by all 4 waves:
// wave 0 DMAs them into double-buffered LDS with tensor_load_to_lds, overlapped
// one tile ahead; completion via s_wait_tensorcnt + block barrier.

__global__ __launch_bounds__(128) void k_flash_attn(
    const bf16* __restrict__ qh, const bf16* __restrict__ kh, const bf16* __restrict__ vt,
    bf16* __restrict__ ao) {
  __shared__ __align__(16) bf16 kbuf[2][32 * 64];   // [key_local][d]
  __shared__ __align__(16) bf16 vbuf[2][64 * 32];   // [d][key_local]
  __shared__ __align__(16) bf16 plds[4][16 * 32];   // per-wave P staging
  int lane = threadIdx.x & 31;
  int w    = threadIdx.x >> 5;
  int qt   = blockIdx.x;
  int h    = blockIdx.y;
  int b    = blockIdx.z;
  int kvh  = h >> 1;                 // GQA rep=2
  int half = lane >> 4, l16 = lane & 15;
  int koff = half * 8;
  int q0   = qt * 64 + w * 16;

  const bf16* qp = qh + ((b * NHEAD + h) * LSEQ + q0) * 64;
  const bf16* kp = kh + ((b * NKV + kvh) * LSEQ) * 64;
  const bf16* vp = vt + (b * NKV + kvh) * 64 * LSEQ;

  const bf16* qr = qp + l16 * 64;
  v16bf aq0 = *(const v16bf*)(qr + half * 16);
  v16bf aq1 = *(const v16bf*)(qr + 32 + half * 16);

  float mrow[8], ssum[8];
  v8f o0 = vzero8(), o1 = vzero8(), o2 = vzero8(), o3 = vzero8();
#pragma unroll
  for (int r = 0; r < 8; ++r) { mrow[r] = -1e30f; ssum[r] = 0.0f; }

  bf16* myp = &plds[w][0];
  const int ntile = LSEQ / 32;

  if (w == 0) {
    tdm_load_tile_bf16(kp, &kbuf[0][0], 64, 32, 64, 32, 64);            // K tile 0
    tdm_load_tile_bf16(vp, &vbuf[0][0], 32, 64, 32, 64, LSEQ);          // V tile 0
  }

  for (int it = 0; it < ntile; ++it) {
    int buf = it & 1;
    if (w == 0) __builtin_amdgcn_s_wait_tensorcnt(0);
    __syncthreads();                                  // tile `it` visible; prev reads done
    if (w == 0 && it + 1 < ntile) {
      int key1 = (it + 1) * 32;
      tdm_load_tile_bf16(kp + key1 * 64, &kbuf[buf ^ 1][0], 64, 32, 64, 32, 64);
      tdm_load_tile_bf16(vp + key1,      &vbuf[buf ^ 1][0], 32, 64, 32, 64, LSEQ);
    }
    const bf16* klds = &kbuf[buf][0];
    const bf16* vlds = &vbuf[buf][0];

    // scores S = Q * K^T : two 16x16 accs (keys 0..15, 16..31 of tile)
    v8f s0 = vzero8(), s1 = vzero8();
    s0 = wmma_bf16(aq0, *(const v16bf*)(klds + ( 0 + l16) * 64 + half * 16),      s0);
    s0 = wmma_bf16(aq1, *(const v16bf*)(klds + ( 0 + l16) * 64 + 32 + half * 16), s0);
    s1 = wmma_bf16(aq0, *(const v16bf*)(klds + (16 + l16) * 64 + half * 16),      s1);
    s1 = wmma_bf16(aq1, *(const v16bf*)(klds + (16 + l16) * 64 + 32 + half * 16), s1);

    float mx[8];
#pragma unroll
    for (int r = 0; r < 8; ++r) {
      s0[r] *= 0.125f; s1[r] *= 0.125f;       // 1/sqrt(64)
      mx[r] = fmaxf(s0[r], s1[r]);
    }
#pragma unroll
    for (int off = 8; off >= 1; off >>= 1)
#pragma unroll
      for (int r = 0; r < 8; ++r)
        mx[r] = fmaxf(mx[r], __shfl_xor(mx[r], off, 16));

    float c[8], rs[8];
#pragma unroll
    for (int r = 0; r < 8; ++r) {
      float mn = fmaxf(mrow[r], mx[r]);
      c[r] = __expf(mrow[r] - mn);
      mrow[r] = mn;
      float p0 = __expf(s0[r] - mn);
      float p1 = __expf(s1[r] - mn);
      s0[r] = p0; s1[r] = p1;
      rs[r] = p0 + p1;
    }
#pragma unroll
    for (int off = 8; off >= 1; off >>= 1)
#pragma unroll
      for (int r = 0; r < 8; ++r)
        rs[r] += __shfl_xor(rs[r], off, 16);
#pragma unroll
    for (int r = 0; r < 8; ++r) {
      ssum[r] = ssum[r] * c[r] + rs[r];
      o0[r] *= c[r]; o1[r] *= c[r]; o2[r] *= c[r]; o3[r] *= c[r];
    }

    // P (C-layout) -> LDS -> reload as A-fragment layout
#pragma unroll
    for (int r = 0; r < 8; ++r) {
      int Mr = r + half * 8;
      myp[Mr * 32 + l16]      = (bf16)s0[r];
      myp[Mr * 32 + 16 + l16] = (bf16)s1[r];
    }
    __syncthreads();
    const bf16* prow = myp + l16 * 32;
    v16bf ap = *(const v16bf*)(prow + half * 16);

    // O += P * V  (V transposed in LDS: rows = head-dim, contiguous over keys)
    o0 = wmma_bf16(ap, *(const v16bf*)(vlds + ( 0 + l16) * 32 + half * 16), o0);
    o1 = wmma_bf16(ap, *(const v16bf*)(vlds + (16 + l16) * 32 + half * 16), o1);
    o2 = wmma_bf16(ap, *(const v16bf*)(vlds + (32 + l16) * 32 + half * 16), o2);
    o3 = wmma_bf16(ap, *(const v16bf*)(vlds + (48 + l16) * 32 + half * 16), o3);
    __syncthreads();
  }

#pragma unroll
  for (int r = 0; r < 8; ++r) {
    float inv = 1.0f / ssum[r];
    int grow = b * LSEQ + q0 + r + half * 8;
    bf16* dst = ao + grow * CDIM + h * 64 + l16;
    dst[0]  = (bf16)(o0[r] * inv);
    dst[16] = (bf16)(o1[r] * inv);
    dst[32] = (bf16)(o2[r] * inv);
    dst[48] = (bf16)(o3[r] * inv);
  }
}

// ---------------- host launch ----------------

extern "C" void kernel_launch(void* const* d_in, const int* in_sizes, int n_in,
                              void* d_out, int out_size, void* d_ws, size_t ws_size,
                              hipStream_t stream) {
  (void)in_sizes; (void)n_in; (void)out_size; (void)ws_size;
  const float* x   = (const float*)d_in[0];
  const float* wq  = (const float*)d_in[1];
  const float* wk  = (const float*)d_in[2];
  const float* wv  = (const float*)d_in[3];
  const float* wo  = (const float*)d_in[4];
  const float* qnw = (const float*)d_in[5];
  const float* knw = (const float*)d_in[6];
  float* out = (float*)d_out;

  char* ws = (char*)d_ws;
  bf16*  xb  = (bf16*)(ws);                    //  4,718,592 B  x in bf16 [4608,512]
  bf16*  wT  = (bf16*)(ws + 4718592);          //  1,572,864 B  [wqT|wkT|wvT|woT], ld=512
  bf16*  wqT = wT;
  bf16*  wkT = wT + 512 * 512;
  bf16*  wvT = wkT + 256 * 512;
  bf16*  woT = wvT + 256 * 512;
  float* qkv = (float*)(ws + 6291456);         // 18,874,368 B  [4608,1024] f32
  bf16*  qhb = (bf16*)(ws + 25165824);         //  4,718,592 B  [B,nH,L,hd]
  bf16*  khb = (bf16*)(ws + 29884416);         //  2,359,296 B  [B,nKV,L,hd]
  bf16*  vtb = (bf16*)(ws + 32243712);         //  2,359,296 B  [B,nKV,hd,L]
  bf16*  aob = (bf16*)(ws + 34603008);         //  4,718,592 B  attn out bf16 [4608,512]

  int nx = NB * LSEQ * CDIM;
  k_cast_bf16<<<(nx + 255) / 256, 256, 0, stream>>>(x, xb, nx);
  k_transpose_bf16<<<(512 * 512 + 255) / 256, 256, 0, stream>>>(wq, wqT, 512, 512);
  k_transpose_bf16<<<(512 * 256 + 255) / 256, 256, 0, stream>>>(wk, wkT, 512, 256);
  k_transpose_bf16<<<(512 * 256 + 255) / 256, 256, 0, stream>>>(wv, wvT, 512, 256);
  k_transpose_bf16<<<(512 * 512 + 255) / 256, 256, 0, stream>>>(wo, woT, 512, 512);

  // fused QKV projection: [4608,512] x [512,1024]
  k_gemm_wmma<<<dim3(1024 / 64, MROWS / 64), 128, 0, stream>>>(xb, wT, qkv, 512, 512, 1024, 512);
  // RMSNorm + RoPE + head-major / transposed-V packing
  k_norm_rope_pack<<<(NB * LSEQ * 16) / 8, 256, 0, stream>>>(qkv, qnw, knw, qhb, khb, vtb);
  // flash attention
  k_flash_attn<<<dim3(LSEQ / 64, NHEAD, NB), 128, 0, stream>>>(qhb, khb, vtb, aob);
  // output projection: [4608,512] x [512,512]
  k_gemm_wmma<<<dim3(512 / 64, MROWS / 64), 128, 0, stream>>>(aob, woT, out, 512, 512, 512, 512);
}
